// SpikingNet_44822278701095
// MI455X (gfx1250) — matrix operand order repaired and checked by reading, not compiled, not run on hardware
//
#include <hip/hip_runtime.h>

// ---------------------------------------------------------------------------
// CDNA5 (gfx1250) spiking-net forward: f16 WMMA GEMMs + fused f32 LIF updates.
// T=50, B=128, NIN=1024, H=2048, NOUT=256.
//
// Roofline: weights in f16 = 61 MB -> resident in 192 MB L2 after first step;
// HBM traffic is a one-time ~75 MB (~3 us @ 23.3 TB/s). The run is bound by
// the 300-deep sequential layer chain + skinny-GEMM latency, so we maximize
// per-wave WMMA density: each wave sweeps K once and produces TWO 16x16
// output tiles (shared A fragment, two independent accumulators).
// ---------------------------------------------------------------------------

typedef __attribute__((ext_vector_type(8)))  _Float16 v8h;
typedef __attribute__((ext_vector_type(16))) _Float16 v16h;
typedef __attribute__((ext_vector_type(8)))  float    v8f;

union AFrag { v16h v; v8h h[2]; };

#define SNN_ALPHA 0.9f
#define SNN_BETA  0.9f
#define SNN_THR   1.0f

// --------------------------- helper kernels --------------------------------

__global__ __launch_bounds__(256) void cvt_f32_to_f16(const float* __restrict__ s,
                                                      _Float16* __restrict__ d, int n) {
  int i = blockIdx.x * 256 + threadIdx.x;
  if (i < n) d[i] = (_Float16)s[i];
}

__global__ __launch_bounds__(256) void zero_f32_kernel(float* __restrict__ p, int n) {
  int i = blockIdx.x * 256 + threadIdx.x;
  if (i < n) p[i] = 0.0f;
}

__global__ __launch_bounds__(256) void zero_f16_kernel(_Float16* __restrict__ p, int n) {
  int i = blockIdx.x * 256 + threadIdx.x;
  if (i < n) p[i] = (_Float16)0.0f;
}

// --------------------------- WMMA dual-tile GEMM ---------------------------
// A fragment (16-bit 16x32): lanes 0-15 hold K 0..15,
// lanes 16-31 hold K 16..31 -> two v8h loads. B = W^T, so a B column
// is a contiguous W row: lanes 0-15 hold K 0..15, lanes 16-31 hold K 16..31
// -> one v16h load per tile. One shared A fragment feeds two W-row streams
// (two N-tiles), giving two independent WMMA accumulator chains per K-step.
__device__ __forceinline__ void gemm_tile_acc2(const _Float16* __restrict__ arow,
                                               const _Float16* __restrict__ wrow0,
                                               const _Float16* __restrict__ wrow1,
                                               int K, int koffA, int koffB,
                                               v8f& acc0, v8f& acc1) {
#pragma unroll 2
  for (int k0 = 0; k0 < K; k0 += 32) {
    // Speculative prefetch a few K-steps ahead on the weight streams
    // (gfx1250 global_prefetch; OOB speculative prefetches are dropped).
    __builtin_prefetch((const void*)(wrow0 + k0 + 256), 0, 1);
    __builtin_prefetch((const void*)(wrow1 + k0 + 256), 0, 1);
    AFrag af;
    af.h[0] = *(const v8h*)(arow + k0 + koffA);
    af.h[1] = *(const v8h*)(arow + k0 + koffA + 8);
    v16h b0 = *(const v16h*)(wrow0 + k0 + koffB);
    v16h b1 = *(const v16h*)(wrow1 + k0 + koffB);
    acc0 = __builtin_amdgcn_wmma_f32_16x16x32_f16(
        false, af.v, false, b0, (short)0, acc0, false, false);
    acc1 = __builtin_amdgcn_wmma_f32_16x16x32_f16(
        false, af.v, false, b1, (short)0, acc1, false, false);
  }
}

// --------------------------- fused LIF epilogue ----------------------------
// C/D layout: VGPR r -> M = r + (lane>=16 ? 8 : 0), N = lane&15.
__device__ __forceinline__ void lif_update(const v8f& acc, float bb, int m0,
                                           int n_col, int N,
                                           float* __restrict__ syn,
                                           float* __restrict__ mem,
                                           _Float16* __restrict__ spk_out,
                                           float* __restrict__ out_spk_f32,
                                           float* __restrict__ out_mem_f32) {
#pragma unroll
  for (int r = 0; r < 8; ++r) {
    const int m = m0 + r;
    const size_t idx = (size_t)m * (size_t)N + (size_t)n_col;
    const float cur = acc[r] + bb;
    float mv = mem[idx];
    const float reset = (mv - SNN_THR > 0.0f) ? 1.0f : 0.0f;
    float spkv;
    if (syn != nullptr) {                  // RSynaptic (reset_delay = False)
      float sv = SNN_ALPHA * syn[idx] + cur;
      mv = SNN_BETA * mv + sv - reset * SNN_THR;
      spkv = (mv - SNN_THR > 0.0f) ? 1.0f : 0.0f;
      mv = mv - (spkv - reset) * SNN_THR;
      syn[idx] = sv;
    } else {                               // Leaky (reset_delay = True)
      mv = SNN_BETA * mv + cur - reset * SNN_THR;
      spkv = (mv - SNN_THR > 0.0f) ? 1.0f : 0.0f;
    }
    mem[idx] = mv;
    if (spk_out != nullptr)     spk_out[idx] = (_Float16)spkv;
    if (out_spk_f32 != nullptr) { out_spk_f32[idx] = spkv; out_mem_f32[idx] = mv; }
  }
}

// --------------------------- fused layer kernel ----------------------------
// Block = 256 threads = 8 waves; wave w = M-tile w (B=128 -> 8 m-tiles).
// Each block owns TWO N-tiles (32 output columns); each wave computes two
// 16x16 tiles sharing its A fragment. Grid = N/32 blocks.
__global__ __launch_bounds__(256) void snn_layer_kernel(
    const _Float16* __restrict__ A, int K1,
    const _Float16* __restrict__ W, const float* __restrict__ bias,
    const _Float16* __restrict__ Aself, const _Float16* __restrict__ Wr,
    const float* __restrict__ brbias, int K2,
    float* __restrict__ syn, float* __restrict__ mem,
    _Float16* __restrict__ spk_out,
    float* __restrict__ out_spk_f32, float* __restrict__ out_mem_f32,
    int N) {
  const int lane  = threadIdx.x & 31;
  const int wave  = threadIdx.x >> 5;
  const int mtile = wave;                      // 8 m-tiles cover B=128
  const int nl    = lane & 15;
  const bool hi   = (lane >= 16);
  const int koffA = hi ? 16 : 0;
  const int koffB = hi ? 16 : 0;
  const int m_row  = mtile * 16 + nl;          // A row this lane streams
  const int n_col0 = blockIdx.x * 32 + nl;     // W row for N-tile 0
  const int n_col1 = n_col0 + 16;              // W row for N-tile 1

  v8f acc0 = {}, acc1 = {};
  gemm_tile_acc2(A + (size_t)m_row * K1,
                 W + (size_t)n_col0 * K1, W + (size_t)n_col1 * K1,
                 K1, koffA, koffB, acc0, acc1);
  if (Wr != nullptr) {
    gemm_tile_acc2(Aself + (size_t)m_row * K2,
                   Wr + (size_t)n_col0 * K2, Wr + (size_t)n_col1 * K2,
                   K2, koffA, koffB, acc0, acc1);
  }

  float bb0 = bias[n_col0];
  float bb1 = bias[n_col1];
  if (brbias != nullptr) { bb0 += brbias[n_col0]; bb1 += brbias[n_col1]; }

  const int m0 = mtile * 16 + (hi ? 8 : 0);
  lif_update(acc0, bb0, m0, n_col0, N, syn, mem, spk_out, out_spk_f32, out_mem_f32);
  lif_update(acc1, bb1, m0, n_col1, N, syn, mem, spk_out, out_spk_f32, out_mem_f32);
}

// ------------------------------- launch ------------------------------------

extern "C" void kernel_launch(void* const* d_in, const int* in_sizes, int n_in,
                              void* d_out, int out_size, void* d_ws, size_t ws_size,
                              hipStream_t stream) {
  (void)in_sizes; (void)n_in; (void)out_size; (void)ws_size;
  const int T = 50, B = 128, NIN = 1024, H = 2048, NOUT = 256;

  const float* x   = (const float*)d_in[0];
  const float* W1  = (const float*)d_in[1];
  const float* b1  = (const float*)d_in[2];
  const float* Wr1 = (const float*)d_in[3];
  const float* br1 = (const float*)d_in[4];
  const float* W2  = (const float*)d_in[5];
  const float* b2  = (const float*)d_in[6];
  const float* Wr2 = (const float*)d_in[7];
  const float* br2 = (const float*)d_in[8];
  const float* W3  = (const float*)d_in[9];
  const float* b3  = (const float*)d_in[10];
  const float* Wr3 = (const float*)d_in[11];
  const float* br3 = (const float*)d_in[12];
  const float* W4  = (const float*)d_in[13];
  const float* b4  = (const float*)d_in[14];
  const float* W5  = (const float*)d_in[15];
  const float* b5  = (const float*)d_in[16];
  const float* W6  = (const float*)d_in[17];
  const float* b6  = (const float*)d_in[18];
  float* out = (float*)d_out;

  // Bump allocator over d_ws (256B aligned blocks).
  char* base = (char*)d_ws;
  size_t off = 0;
  auto alloc = [&](size_t bytes) -> void* {
    void* p = base + off;
    off = (off + bytes + 255) & ~(size_t)255;
    return p;
  };

  const size_t BH = (size_t)B * H;
  _Float16* xh   = (_Float16*)alloc((size_t)T * B * NIN * 2);
  _Float16* w1h  = (_Float16*)alloc((size_t)H * NIN * 2);
  _Float16* wr1h = (_Float16*)alloc((size_t)H * H * 2);
  _Float16* w2h  = (_Float16*)alloc((size_t)H * H * 2);
  _Float16* wr2h = (_Float16*)alloc((size_t)H * H * 2);
  _Float16* w3h  = (_Float16*)alloc((size_t)H * H * 2);
  _Float16* wr3h = (_Float16*)alloc((size_t)H * H * 2);
  _Float16* w4h  = (_Float16*)alloc((size_t)H * H * 2);
  _Float16* w5h  = (_Float16*)alloc((size_t)H * H * 2);
  _Float16* w6h  = (_Float16*)alloc((size_t)NOUT * H * 2);
  // Spike buffers (contiguous so one zero launch clears them all).
  _Float16* spkbuf = (_Float16*)alloc(8 * BH * 2);   // spk1[2],spk2[2],spk3[2],spk4,spk5
  _Float16* spk1[2] = { spkbuf + 0 * BH, spkbuf + 1 * BH };
  _Float16* spk2[2] = { spkbuf + 2 * BH, spkbuf + 3 * BH };
  _Float16* spk3[2] = { spkbuf + 4 * BH, spkbuf + 5 * BH };
  _Float16* spk4 = spkbuf + 6 * BH;
  _Float16* spk5 = spkbuf + 7 * BH;
  // f32 states (contiguous).
  const size_t nStateF32 = 8 * BH + (size_t)B * NOUT;
  float* statef = (float*)alloc(nStateF32 * 4);
  float* syn1 = statef + 0 * BH;
  float* mem1 = statef + 1 * BH;
  float* syn2 = statef + 2 * BH;
  float* mem2 = statef + 3 * BH;
  float* syn3 = statef + 4 * BH;
  float* mem3 = statef + 5 * BH;
  float* mem4 = statef + 6 * BH;
  float* mem5 = statef + 7 * BH;
  float* mem6 = statef + 8 * BH;

  auto cvt = [&](const float* s, _Float16* d, size_t n) {
    cvt_f32_to_f16<<<(unsigned)((n + 255) / 256), 256, 0, stream>>>(s, d, (int)n);
  };
  cvt(x,   xh,   (size_t)T * B * NIN);
  cvt(W1,  w1h,  (size_t)H * NIN);
  cvt(Wr1, wr1h, (size_t)H * H);
  cvt(W2,  w2h,  (size_t)H * H);
  cvt(Wr2, wr2h, (size_t)H * H);
  cvt(W3,  w3h,  (size_t)H * H);
  cvt(Wr3, wr3h, (size_t)H * H);
  cvt(W4,  w4h,  (size_t)H * H);
  cvt(W5,  w5h,  (size_t)H * H);
  cvt(W6,  w6h,  (size_t)NOUT * H);

  zero_f16_kernel<<<(unsigned)((8 * BH + 255) / 256), 256, 0, stream>>>(spkbuf, (int)(8 * BH));
  zero_f32_kernel<<<(unsigned)((nStateF32 + 255) / 256), 256, 0, stream>>>(statef, (int)nStateF32);

  const dim3 blk(256);
  const dim3 gridH(H / 32);      // 64 blocks * 8 waves * 2 tiles = 1024 tiles
  const dim3 gridO(NOUT / 32);   // 8 blocks

  for (int t = 0; t < T; ++t) {
    const int pr = t & 1;        // read own spikes from previous step
    const int pw = (t + 1) & 1;  // write this step's spikes
    const _Float16* xt = xh + (size_t)t * B * NIN;

    // Layer 1: RSynaptic, input current from x.
    snn_layer_kernel<<<gridH, blk, 0, stream>>>(
        xt, NIN, w1h, b1, spk1[pr], wr1h, br1, H,
        syn1, mem1, spk1[pw], nullptr, nullptr, H);
    // Layer 2: RSynaptic.
    snn_layer_kernel<<<gridH, blk, 0, stream>>>(
        spk1[pw], H, w2h, b2, spk2[pr], wr2h, br2, H,
        syn2, mem2, spk2[pw], nullptr, nullptr, H);
    // Layer 3: RSynaptic.
    snn_layer_kernel<<<gridH, blk, 0, stream>>>(
        spk2[pw], H, w3h, b3, spk3[pr], wr3h, br3, H,
        syn3, mem3, spk3[pw], nullptr, nullptr, H);
    // Layer 4: Leaky.
    snn_layer_kernel<<<gridH, blk, 0, stream>>>(
        spk3[pw], H, w4h, b4, nullptr, nullptr, nullptr, 0,
        nullptr, mem4, spk4, nullptr, nullptr, H);
    // Layer 5: Leaky.
    snn_layer_kernel<<<gridH, blk, 0, stream>>>(
        spk4, H, w5h, b5, nullptr, nullptr, nullptr, 0,
        nullptr, mem5, spk5, nullptr, nullptr, H);
    // Layer 6: Leaky output -> d_out (spk_rec then mem_rec, flat).
    float* out_s = out + (size_t)t * B * NOUT;
    float* out_m = out + (size_t)T * B * NOUT + (size_t)t * B * NOUT;
    snn_layer_kernel<<<gridO, blk, 0, stream>>>(
        spk5, H, w6h, b6, nullptr, nullptr, nullptr, 0,
        nullptr, mem6, nullptr, out_s, out_m, NOUT);
  }
}
